// BitLinearKernel_32478542693200
// MI455X (gfx1250) — 8-bit matrix operand order repaired and checked by reading, compiled, not run on hardware
//
#include <hip/hip_runtime.h>

typedef __attribute__((ext_vector_type(8))) int v8i;

#if defined(__has_builtin)
#if __has_builtin(__builtin_amdgcn_sched_barrier)
#define SCHED_FENCE() __builtin_amdgcn_sched_barrier(0)
#endif
#endif
#ifndef SCHED_FENCE
#define SCHED_FENCE()
#endif

constexpr int Bb = 2, Ss = 2048, Kk = 4096, Nn = 16384;
constexpr int M  = Bb * Ss;
constexpr int KB = Kk / 4;

__device__ __forceinline__ size_t tiled_off(int outer, int k) {
  return ((size_t)((outer >> 7) * 64 + (k >> 6)) << 13) +
         ((outer & 127) << 6) + (k & 63);
}

__device__ __forceinline__ unsigned int expand_tern(unsigned int b) {
  unsigned int r = (b & 3u) | ((b & 0x0Cu) << 6) | ((b & 0x30u) << 12) |
                   ((b & 0xC0u) << 18);
  r |= ((r >> 1) & 0x01010101u) * 0xFCu;
  return r;
}

__global__ __launch_bounds__(256) void bitlinear_expand(
    const unsigned int* __restrict__ wp32, unsigned char* __restrict__ wx) {
  const int i  = blockIdx.x * 256 + threadIdx.x;
  const int n  = i >> 8;
  const int kw = i & 255;
  const unsigned int w = wp32[i];
  uint4 o;
  o.x = expand_tern(w & 0xFF);
  o.y = expand_tern((w >> 8) & 0xFF);
  o.z = expand_tern((w >> 16) & 0xFF);
  o.w = expand_tern(w >> 24);
  *(uint4*)(wx + tiled_off(n, kw * 16)) = o;
}

__global__ __launch_bounds__(256) void bitlinear_quant(
    const float* __restrict__ x, unsigned char* __restrict__ xq,
    float* __restrict__ inv_s) {
  __shared__ float red[256];
  const int row = blockIdx.x;
  const int t   = threadIdx.x;

  const float4* xr = (const float4*)(x + (size_t)row * Kk);
  float4 v[4];
  float amax = 0.0f;
#pragma unroll
  for (int i = 0; i < 4; ++i) {
    v[i] = xr[t * 4 + i];
    amax = fmaxf(amax, fmaxf(fmaxf(fabsf(v[i].x), fabsf(v[i].y)),
                             fmaxf(fabsf(v[i].z), fabsf(v[i].w))));
  }
  red[t] = amax;
  __syncthreads();
  for (int off = 128; off > 0; off >>= 1) {
    if (t < off) red[t] = fmaxf(red[t], red[t + off]);
    __syncthreads();
  }
  const float am = fmaxf(red[0], 1e-5f);
  const float s  = 127.0f / am;
  if (t == 0) inv_s[row] = am * (1.0f / 127.0f);

  uint4 o;
  unsigned int* op = (unsigned int*)&o;
#pragma unroll
  for (int i = 0; i < 4; ++i) {
    const float4 q = v[i];
    const int a = (int)fminf(fmaxf(rintf(q.x * s), -128.0f), 127.0f);
    const int b = (int)fminf(fmaxf(rintf(q.y * s), -128.0f), 127.0f);
    const int c = (int)fminf(fmaxf(rintf(q.z * s), -128.0f), 127.0f);
    const int d = (int)fminf(fmaxf(rintf(q.w * s), -128.0f), 127.0f);
    op[i] = (a & 0xFF) | ((b & 0xFF) << 8) | ((c & 0xFF) << 16) |
            ((unsigned)(d & 0xFF) << 24);
  }
  *(uint4*)(xq + tiled_off(row, t * 16)) = o;
}

__device__ __forceinline__ v8i load_afrag(const unsigned char* p) {
  uint4 p0 = *(const uint4*)(p);
  uint4 p1 = *(const uint4*)(p + 32);
  v8i a;
  a[0] = (int)p0.x; a[1] = (int)p0.y;
  a[2] = (int)p0.z; a[3] = (int)p0.w;
  a[4] = (int)p1.x; a[5] = (int)p1.y;
  a[6] = (int)p1.z; a[7] = (int)p1.w;
  return a;
}
__device__ __forceinline__ v8i load_bfrag(const unsigned char* p) {
  uint4 q0 = *(const uint4*)(p);
  uint4 q1 = *(const uint4*)(p + 32);
  v8i b;
  b[0] = (int)q0.x; b[1] = (int)q0.y; b[2] = (int)q0.z; b[3] = (int)q0.w;
  b[4] = (int)q1.x; b[5] = (int)q1.y; b[6] = (int)q1.z; b[7] = (int)q1.w;
  return b;
}

__global__ __launch_bounds__(256) void bitlinear_gemm(
    const unsigned char* __restrict__ xq, const float* __restrict__ inv_s,
    const unsigned char* __restrict__ wx, const float* __restrict__ wscale,
    float* __restrict__ out) {
  constexpr int LDT  = 80;
  constexpr int BUFB = 128 * LDT;
  __shared__ unsigned char lds_a[2 * BUFB];
  __shared__ unsigned char lds_b[2 * BUFB];

  const int t    = threadIdx.x;
  const int lane = t & 31;
  const int wave = t >> 5;
  const int wm   = wave >> 2;
  const int wn   = wave & 3;
  const int h    = lane >> 4;
  const int ln   = lane & 15;

  const int m0 = blockIdx.y * 128;
  const int n0 = blockIdx.x * 128;

  const unsigned char* a_pan = xq + ((size_t)(m0 >> 7) << 19);
  const unsigned char* b_pan = wx + ((size_t)(n0 >> 7) << 19);

  const int s_off = (t >> 1) * LDT + (t & 1) * 32;

  const int a_base = (wm * 64 + ln) * LDT + h * 16;
  const int b_base = (wn * 32 + ln) * LDT + h * 16;

  v8i acc[4][2] = {};

  const unsigned char* rd_a = lds_a;
  const unsigned char* rd_b = lds_b;
  unsigned char*       st_a = lds_a + BUFB;
  unsigned char*       st_b = lds_b + BUFB;

  {
    const uint4* as = (const uint4*)(a_pan + t * 32);
    const uint4* bs = (const uint4*)(b_pan + t * 32);
    uint4 a0 = as[0], a1 = as[1], b0 = bs[0], b1 = bs[1];
    *(uint4*)&lds_a[s_off]      = a0;
    *(uint4*)&lds_a[s_off + 16] = a1;
    *(uint4*)&lds_b[s_off]      = b0;
    *(uint4*)&lds_b[s_off + 16] = b1;
  }
  __syncthreads();

#pragma clang loop unroll(disable)
  for (int kt = 0; kt < 63; ++kt) {
    const uint4* as = (const uint4*)(a_pan + ((size_t)(kt + 1) << 13) + t * 32);
    const uint4* bs = (const uint4*)(b_pan + ((size_t)(kt + 1) << 13) + t * 32);
    uint4 ga0 = as[0], ga1 = as[1], gb0 = bs[0], gb1 = bs[1];

    v8i bf0 = load_bfrag(rd_b + b_base);
    v8i bf1 = load_bfrag(rd_b + b_base + 16 * LDT);
    v8i af0 = load_afrag(rd_a + a_base);
    v8i af1 = load_afrag(rd_a + a_base + 16 * LDT);
    v8i af2 = load_afrag(rd_a + a_base + 32 * LDT);
    v8i af3 = load_afrag(rd_a + a_base + 48 * LDT);

    *(uint4*)&st_a[s_off]      = ga0;
    *(uint4*)&st_a[s_off + 16] = ga1;
    *(uint4*)&st_b[s_off]      = gb0;
    *(uint4*)&st_b[s_off + 16] = gb1;

    acc[0][0] = __builtin_amdgcn_wmma_i32_16x16x64_iu8(true, af0, true, bf0,
                                                       acc[0][0], false, false);
    acc[0][1] = __builtin_amdgcn_wmma_i32_16x16x64_iu8(true, af0, true, bf1,
                                                       acc[0][1], false, false);
    acc[1][0] = __builtin_amdgcn_wmma_i32_16x16x64_iu8(true, af1, true, bf0,
                                                       acc[1][0], false, false);
    acc[1][1] = __builtin_amdgcn_wmma_i32_16x16x64_iu8(true, af1, true, bf1,
                                                       acc[1][1], false, false);
    acc[2][0] = __builtin_amdgcn_wmma_i32_16x16x64_iu8(true, af2, true, bf0,
                                                       acc[2][0], false, false);
    acc[2][1] = __builtin_amdgcn_wmma_i32_16x16x64_iu8(true, af2, true, bf1,
                                                       acc[2][1], false, false);
    acc[3][0] = __builtin_amdgcn_wmma_i32_16x16x64_iu8(true, af3, true, bf0,
                                                       acc[3][0], false, false);
    acc[3][1] = __builtin_amdgcn_wmma_i32_16x16x64_iu8(true, af3, true, bf1,
                                                       acc[3][1], false, false);

    const unsigned char* ta = rd_a; rd_a = st_a; st_a = (unsigned char*)ta;
    const unsigned char* tb = rd_b; rd_b = st_b; st_b = (unsigned char*)tb;
    SCHED_FENCE();
    __syncthreads();
  }

  {
    v8i bf0 = load_bfrag(rd_b + b_base);
    v8i bf1 = load_bfrag(rd_b + b_base + 16 * LDT);
    v8i af0 = load_afrag(rd_a + a_base);
    v8i af1 = load_afrag(rd_a + a_base + 16 * LDT);
    v8i af2 = load_afrag(rd_a + a_base + 32 * LDT);
    v8i af3 = load_afrag(rd_a + a_base + 48 * LDT);
    acc[0][0] = __builtin_amdgcn_wmma_i32_16x16x64_iu8(true, af0, true, bf0,
                                                       acc[0][0], false, false);
    acc[0][1] = __builtin_amdgcn_wmma_i32_16x16x64_iu8(true, af0, true, bf1,
                                                       acc[0][1], false, false);
    acc[1][0] = __builtin_amdgcn_wmma_i32_16x16x64_iu8(true, af1, true, bf0,
                                                       acc[1][0], false, false);
    acc[1][1] = __builtin_amdgcn_wmma_i32_16x16x64_iu8(true, af1, true, bf1,
                                                       acc[1][1], false, false);
    acc[2][0] = __builtin_amdgcn_wmma_i32_16x16x64_iu8(true, af2, true, bf0,
                                                       acc[2][0], false, false);
    acc[2][1] = __builtin_amdgcn_wmma_i32_16x16x64_iu8(true, af2, true, bf1,
                                                       acc[2][1], false, false);
    acc[3][0] = __builtin_amdgcn_wmma_i32_16x16x64_iu8(true, af3, true, bf0,
                                                       acc[3][0], false, false);
    acc[3][1] = __builtin_amdgcn_wmma_i32_16x16x64_iu8(true, af3, true, bf1,
                                                       acc[3][1], false, false);
  }

  const float ws0 = wscale[0];
#pragma unroll
  for (int tm = 0; tm < 4; ++tm) {
    const int mbase = m0 + wm * 64 + tm * 16 + h * 8;
#pragma unroll
    for (int j = 0; j < 8; ++j) {
      const float  sc = ws0 * inv_s[mbase + j];
      const size_t orow = (size_t)(mbase + j) * Nn;
      out[orow + n0 + wn * 32 + ln]      = (float)acc[tm][0][j] * sc;
      out[orow + n0 + wn * 32 + 16 + ln] = (float)acc[tm][1][j] * sc;
    }
  }
}

extern "C" void kernel_launch(void* const* d_in, const int* in_sizes, int n_in,
                              void* d_out, int out_size, void* d_ws,
                              size_t ws_size, hipStream_t stream) {
  const float*        x      = (const float*)d_in[0];
  const unsigned int* wp32   = (const unsigned int*)d_in[1];
  const float*        wscale = (const float*)d_in[2];
  float*              out    = (float*)d_out;

  unsigned char* wx    = (unsigned char*)d_ws;
  unsigned char* xq    = (unsigned char*)d_ws + (size_t)Nn * Kk;
  float*         inv_s = (float*)((unsigned char*)d_ws + (size_t)Nn * Kk +
                                  (size_t)M * Kk);

  bitlinear_expand<<<(Nn * (KB / 4)) / 256, 256, 0, stream>>>(wp32, wx);
  bitlinear_quant<<<M, 256, 0, stream>>>(x, xq, inv_s);

  dim3 grid(Nn / 128, M / 128);
  bitlinear_gemm<<<grid, 256, 0, stream>>>(xq, inv_s, wx, wscale, out);
}
